// AMMLinear_15350213116639
// MI455X (gfx1250) — 8-bit matrix operand order repaired and checked by reading, compiled, not run on hardware
//
#include <hip/hip_runtime.h>
#include <hip/hip_bf16.h>
#include <stdint.h>

#define NC    64
#define KC    16
#define SUBV  64
#define INF   4096
#define OUTF  4096
#define NTOK  4096
#define CKDIM 1024

typedef __attribute__((ext_vector_type(2))) float v2f;
typedef __attribute__((ext_vector_type(8))) float v8f;
typedef __attribute__((ext_vector_type(8))) int   v8i;
typedef __attribute__((ext_vector_type(4))) int   v4i;

__global__ void lut_gemm_kernel(const float* __restrict__ centroids,
                                const float* __restrict__ weight,
                                float* __restrict__ lut,
                                unsigned int* __restrict__ maxabs_bits) {
    const int lane  = threadIdx.x & 31;
    const int wave  = threadIdx.x >> 5;
    const int otile = blockIdx.x * 8 + wave;
    const int c     = blockIdx.y;
    const int obase = otile * 16;
    const int m     = lane & 15;
    const int hi    = lane >> 4;

    const float* A  = centroids + (size_t)c * (KC * SUBV);
    const float* Bm = weight    + (size_t)c * (SUBV * OUTF);

    v8f acc = {};
#pragma unroll
    for (int kb = 0; kb < 16; ++kb) {
        const int k0 = kb * 4 + 2 * hi;
        v2f a, b;
        a.x = A[m * SUBV + k0 + 0];
        a.y = A[m * SUBV + k0 + 1];
        b.x = Bm[(size_t)(k0 + 0) * OUTF + obase + m];
        b.y = Bm[(size_t)(k0 + 1) * OUTF + obase + m];
        acc = __builtin_amdgcn_wmma_f32_16x16x4_f32(false, a, false, b,
                                                    (short)0, acc, false, false);
    }

    float mloc = 0.0f;
#pragma unroll
    for (int r = 0; r < 8; ++r) {
        const int krow = r + 8 * hi;
        const float v = acc[r];
        lut[((size_t)c * KC + krow) * OUTF + obase + m] = v;
        mloc = fmaxf(mloc, fabsf(v));
    }
    atomicMax(maxabs_bits, __float_as_uint(mloc));
}

__global__ void quantize_kernel(const float* __restrict__ lut,
                                const unsigned int* __restrict__ maxabs_bits,
                                int8_t* __restrict__ qT) {
    const float maxab = __uint_as_float(*maxabs_bits);
    const float s     = maxab / 127.0f;
    const float inv_s = (s > 0.0f) ? (1.0f / s) : 0.0f;

    const int idx = blockIdx.x * blockDim.x + threadIdx.x;
    const int o   = idx & (OUTF - 1);
    const int ck  = idx >> 12;
    float v = lut[(size_t)ck * OUTF + o] * inv_s;
    v = fminf(fmaxf(v, -128.0f), 127.0f);
    const int q = (int)rintf(v);
    qT[(size_t)o * CKDIM + ck] = (int8_t)q;
}

__global__ void argmin_kernel(const float* __restrict__ x,
                              const float* __restrict__ centroids,
                              uint8_t* __restrict__ idxT) {
    __shared__ float cent[KC * SUBV];
    __shared__ float cnorm[KC];
    const int c   = blockIdx.x;
    const int tid = threadIdx.x;

    for (int i = tid; i < KC * SUBV; i += blockDim.x)
        cent[i] = centroids[(size_t)c * KC * SUBV + i];
    __syncthreads();
    if (tid < KC) {
        float s2 = 0.0f;
        for (int v = 0; v < SUBV; ++v) { float t = cent[tid * SUBV + v]; s2 += t * t; }
        cnorm[tid] = s2;
    }
    __syncthreads();

    const int n = blockIdx.y * blockDim.x + tid;
    const float* xs = x + (size_t)n * INF + c * SUBV;
    float4 xr[16];
#pragma unroll
    for (int i = 0; i < 16; ++i) xr[i] = ((const float4*)xs)[i];

    int   best  = 0;
    float bestd = 3.4e38f;
#pragma unroll 1
    for (int k = 0; k < KC; ++k) {
        const float4* cp = (const float4*)(cent + k * SUBV);
        float dot = 0.0f;
#pragma unroll
        for (int i = 0; i < 16; ++i) {
            float4 cv = cp[i];
            dot += xr[i].x * cv.x + xr[i].y * cv.y + xr[i].z * cv.z + xr[i].w * cv.w;
        }
        const float d = cnorm[k] - 2.0f * dot;
        if (d < bestd) { bestd = d; best = k; }
    }
    idxT[(size_t)n * NC + c] = (uint8_t)best;
}

__global__ void amm_gemm_kernel(const uint8_t* __restrict__ idxT,
                                const int8_t* __restrict__ qT,
                                const unsigned int* __restrict__ maxabs_bits,
                                const float* __restrict__ bias,
                                float* __restrict__ out) {
    const int lane   = threadIdx.x & 31;
    const int wave   = threadIdx.x >> 5;
    const int ogroup = blockIdx.x * 8 + wave;
    const int ntile  = blockIdx.y;
    const int row16  = lane & 15;
    const int sel    = lane >> 4;
    const float s    = __uint_as_float(*maxabs_bits) / 127.0f;

    const int n0  = ntile * 16;
    const int ob0 = ogroup * 128;

    v8i acc[8] = {};

#pragma unroll 1
    for (int kk = 0; kk < 16; ++kk) {
        const unsigned int idx4 =
            *(const unsigned int*)(idxT + (size_t)(n0 + row16) * NC + kk * 4);
        unsigned int areg[8] = {0u, 0u, 0u, 0u, 0u, 0u, 0u, 0u};
        const int halfBase = sel * 8;
#pragma unroll
        for (int j = 0; j < 4; ++j) {
            const int id    = (idx4 >> (8 * j)) & 0xFF;
            if ((j & 1) == sel)
                areg[4 * (j >> 1) + (id >> 2)] = 1u << (8 * (id & 3));
        }
        v8i A;
#pragma unroll
        for (int i = 0; i < 8; ++i) A[i] = (int)areg[i];

#pragma unroll
        for (int t = 0; t < 8; ++t) {
            const int col = ob0 + t * 16 + row16;
            const int8_t* bp = qT + (size_t)col * CKDIM + kk * 64 + sel * 16;
            const v4i blo = *(const v4i*)bp;
            const v4i bhi = *(const v4i*)(bp + 32);
            v8i B;
            B[0] = blo[0]; B[1] = blo[1]; B[2] = blo[2]; B[3] = blo[3];
            B[4] = bhi[0]; B[5] = bhi[1]; B[6] = bhi[2]; B[7] = bhi[3];
            acc[t] = __builtin_amdgcn_wmma_i32_16x16x64_iu8(
                 false, A,  true, B, acc[t], false, false);
        }
    }

#pragma unroll
    for (int t = 0; t < 8; ++t) {
#pragma unroll
        for (int r = 0; r < 8; ++r) {
            const int row = n0 + r + 8 * sel;
            const int col = ob0 + t * 16 + row16;
            out[(size_t)row * OUTF + col] = s * (float)acc[t][r] + bias[col];
        }
    }
}

extern "C" void kernel_launch(void* const* d_in, const int* in_sizes, int n_in,
                              void* d_out, int out_size, void* d_ws, size_t ws_size,
                              hipStream_t stream) {
    const float* x         = (const float*)d_in[0];
    const float* centroids = (const float*)d_in[1];
    const float* weight    = (const float*)d_in[2];
    const float* bias      = (const float*)d_in[4];
    float* out             = (float*)d_out;

    uint8_t* ws = (uint8_t*)d_ws;
    float*        lut    = (float*)ws;
    int8_t*       qT     = (int8_t*)(ws + (size_t)16 * 1024 * 1024);
    uint8_t*      idxT   = (uint8_t*)(ws + (size_t)20 * 1024 * 1024);
    unsigned int* maxabs = (unsigned int*)(ws + (size_t)20 * 1024 * 1024 + 256 * 1024);

    hipMemsetAsync(maxabs, 0, sizeof(unsigned int), stream);

    lut_gemm_kernel<<<dim3(32, 64), 256, 0, stream>>>(centroids, weight, lut, maxabs);
    quantize_kernel<<<(CKDIM * OUTF) / 256, 256, 0, stream>>>(lut, maxabs, qT);
    argmin_kernel<<<dim3(NC, NTOK / 256), 256, 0, stream>>>(x, centroids, idxT);
    amm_gemm_kernel<<<dim3(4, 256), 256, 0, stream>>>(idxT, qT, maxabs, bias, out);
}
